// UltimateDeltaNetDiTBlock_69569880261168
// MI455X (gfx1250) — matrix operand order repaired and checked by reading, not compiled, not run on hardware
//
#include <hip/hip_runtime.h>
#include <math.h>

// ---------------- types ----------------
typedef __bf16 bf16_t;
typedef __attribute__((ext_vector_type(16))) __bf16 v16bf;
typedef __attribute__((ext_vector_type(8)))  float  v8f;

// Problem constants (B=2, N=1024, C=1024, H=16, D=64, P=5, HID=1024)
#define BB 2
#define NN 1024
#define CC 1024
#define HH 16
#define DD 64
#define PP 5
#define TOKS (BB*NN)            // 2048
#define OUT_ELEMS (TOKS*CC)     // 2097152

// ---------------- helpers ----------------
// native conversions -> v_cvt_pk_bf16_f32 on gfx1250
__device__ __forceinline__ unsigned pk_bf16(float lo, float hi) {
  union { bf16_t b[2]; unsigned u; } r;
  r.b[0] = (bf16_t)lo; r.b[1] = (bf16_t)hi;
  return r.u;
}
__device__ __forceinline__ float sigmoidf_(float x) { return 1.0f / (1.0f + __expf(-x)); }
__device__ __forceinline__ float geluf_(float x) {
  return 0.5f * x * (1.0f + tanhf(0.7978845608f * (x + 0.044715f * x * x * x)));
}

// A fragment (16x32 bf16): lane = M (lane&15), half = lane>>4.
// VGPR p: p<4 -> K = (half?8:0)+2p ; p>=4 -> K = (half?24:16)+2(p-4)  (ISA 7.12.2)
__device__ __forceinline__ v16bf load_fragA(const bf16_t* rowp, int half) {
  union { v16bf v; unsigned u[8]; } r;
  const unsigned* p = (const unsigned*)rowp;      // row-contiguous in K, 4B aligned
  #pragma unroll
  for (int i = 0; i < 8; ++i) {
    int kb = (i < 4) ? ((half ? 8 : 0) + 2*i) : ((half ? 24 : 16) + 2*(i-4));
    r.u[i] = p[kb >> 1];
  }
  return r.v;
}
// B fragment (32x16 bf16): lane = N (lane&15); VGPR j: same K order as A:
// j<4 -> K = (half?8:0)+2j ; j>=4 -> K = (half?24:16)+2(j-4). Bs stored transposed [n][k].
__device__ __forceinline__ v16bf load_fragB(const bf16_t* rowp, int half) {
  union { v16bf v; unsigned u[8]; } r;
  const unsigned* p = (const unsigned*)rowp;
  #pragma unroll
  for (int i = 0; i < 8; ++i) {
    int kb = (i < 4) ? ((half ? 8 : 0) + 2*i) : ((half ? 24 : 16) + 2*(i-4));
    r.u[i] = p[kb >> 1];
  }
  return r.v;
}

// ---------------- WMMA GEMM: C[M,N] = act(A[M,K] @ B[K,N] + bias) ----------------
// Block tile 64x64, BK=32, 128 threads = 4 waves, each wave does 2x2 WMMA tiles.
// Double-buffered LDS, one barrier per K-step; B staged as (n, 4k) per thread so
// the transposed store is two packed dwords (no b16 scatter).
__global__ __launch_bounds__(128) void wmma_gemm(
    const float* __restrict__ A, const float* __restrict__ Bm,
    const float* __restrict__ bias, float* __restrict__ C,
    int M, int N, int K, int act) {
  __shared__ bf16_t As[2][64][36];   // [m][k], padded (72B row stride)
  __shared__ bf16_t Bs[2][64][36];   // [n][k], padded (transposed on load)
  const int t    = threadIdx.x;
  const int wave = t >> 5, lane = t & 31;
  const int half = lane >> 4, l16 = lane & 15;
  const int wm = (wave >> 1) * 32, wn = (wave & 1) * 32;
  const int rowBase = blockIdx.y * 64;
  const int colBase = blockIdx.x * 64;

  v8f acc00 = {}, acc01 = {}, acc10 = {}, acc11 = {};
  float4 ra[4];
  float  rb[4][4];

  auto gload = [&](int kb) {
    #pragma unroll
    for (int i = 0; i < 4; ++i) {
      int f = t + 128 * i;
      int r = f >> 3, kq = (f & 7) << 2;              // A tile 64x32, b128 loads
      ra[i] = *(const float4*)(A + (size_t)(rowBase + r) * K + kb + kq);
      int n = f & 63, k4 = (f >> 6) << 2;             // B tile 32x64, 4 coalesced b32
      #pragma unroll
      for (int j = 0; j < 4; ++j)
        rb[i][j] = Bm[(size_t)(kb + k4 + j) * N + colBase + n];
    }
  };
  auto lstore = [&](int buf) {
    #pragma unroll
    for (int i = 0; i < 4; ++i) {
      int f = t + 128 * i;
      int r = f >> 3, kq = (f & 7) << 2;
      *(unsigned*)&As[buf][r][kq]     = pk_bf16(ra[i].x, ra[i].y);
      *(unsigned*)&As[buf][r][kq + 2] = pk_bf16(ra[i].z, ra[i].w);
      int n = f & 63, k4 = (f >> 6) << 2;             // contiguous-in-k store
      *(unsigned*)&Bs[buf][n][k4]     = pk_bf16(rb[i][0], rb[i][1]);
      *(unsigned*)&Bs[buf][n][k4 + 2] = pk_bf16(rb[i][2], rb[i][3]);
    }
  };

  gload(0);
  lstore(0);
  __syncthreads();

  for (int kb = 0; kb < K; kb += 32) {
    const int cur = (kb >> 5) & 1;
    const bool more = (kb + 32) < K;
    if (more) gload(kb + 32);                          // overlaps compute below

    v16bf a0 = load_fragA(&As[cur][wm      + l16][0], half);
    v16bf a1 = load_fragA(&As[cur][wm + 16 + l16][0], half);
    v16bf b0 = load_fragB(&Bs[cur][wn      + l16][0], half);
    v16bf b1 = load_fragB(&Bs[cur][wn + 16 + l16][0], half);
    acc00 = __builtin_amdgcn_wmma_f32_16x16x32_bf16(false, a0, false, b0, (short)0, acc00, false, false);
    acc01 = __builtin_amdgcn_wmma_f32_16x16x32_bf16(false, a0, false, b1, (short)0, acc01, false, false);
    acc10 = __builtin_amdgcn_wmma_f32_16x16x32_bf16(false, a1, false, b0, (short)0, acc10, false, false);
    acc11 = __builtin_amdgcn_wmma_f32_16x16x32_bf16(false, a1, false, b1, (short)0, acc11, false, false);

    if (more) lstore(cur ^ 1);                         // fill next buffer
    __syncthreads();                                   // one barrier per K-step
  }

  // epilogue: C/D layout — VGPR r: lanes0-15 M=r, lanes16-31 M=r+8; N=lane&15
  auto store_acc = [&](v8f acc, int mt, int nt) {
    #pragma unroll
    for (int r = 0; r < 8; ++r) {
      int row = rowBase + wm + mt * 16 + r + half * 8;
      int col = colBase + wn + nt * 16 + l16;
      float v = acc[r];
      if (bias) v += bias[col];
      if (act == 1) v = geluf_(v);
      C[(size_t)row * N + col] = v;
    }
  };
  store_acc(acc00, 0, 0); store_acc(acc01, 0, 1);
  store_acc(acc10, 1, 0); store_acc(acc11, 1, 1);
}

// ---------------- small GEMM (N in {16,80}): wave-per-output dot ----------------
__global__ __launch_bounds__(256) void small_gemm(
    const float* __restrict__ A, const float* __restrict__ W,
    const float* __restrict__ bias, float* __restrict__ O,
    int M, int K, int Nn) {
  int wave = (blockIdx.x * (blockDim.x >> 5)) + (threadIdx.x >> 5);
  int lane = threadIdx.x & 31;
  if (wave >= M * Nn) return;
  int row = wave / Nn, col = wave - row * Nn;
  float acc = 0.f;
  for (int k = lane; k < K; k += 32)
    acc = fmaf(A[(size_t)row * K + k], W[(size_t)k * Nn + col], acc);
  #pragma unroll
  for (int off = 16; off > 0; off >>= 1) acc += __shfl_xor(acc, off, 32);
  if (lane == 0) {
    if (bias) acc += bias[col];
    O[(size_t)row * Nn + col] = acc;
  }
}

// ---------------- delta rule recurrence (beta == 1 identically) ----------------
// One block per (b,h); 512 threads. K-dim split 8-ways across lanes: lane group
// (t&7) owns 8 state rows in registers; k.S / q.S reduced with 3 in-wave
// shfl_xor hops (8 sharers are lane-consecutive -> same wave32). 8 tokens of
// q/k/v staged per LDS round to amortize barriers.
#define TOK_BLK 8
__global__ __launch_bounds__(512) void delta_kernel(
    const float* __restrict__ qkv, float* __restrict__ dout) {
  const int bh = blockIdx.x;
  const int b = bh >> 4, h = bh & 15;
  const int t = threadIdx.x;               // 512
  const int v = t >> 3;                    // 0..63: owned output column
  const int sub = t & 7;                   // 0..7: K-dim slice
  const int kk0 = sub * 8;
  __shared__ float kqv[TOK_BLK][3][64];    // [tok][q|k|v][d]
  float S[8];                              // S[kk0+i][v]
  #pragma unroll
  for (int i = 0; i < 8; ++i) S[i] = 0.f;

  for (int n0 = 0; n0 < NN; n0 += TOK_BLK) {
    // stage 8 tokens (8*192 floats, 512 threads -> 3 each)
    #pragma unroll
    for (int j = 0; j < 3; ++j) {
      int li = t + 512 * j;
      int it = li / 192, w = li % 192;
      int s = w / 64, d = w % 64;
      float val = qkv[(size_t)(b * NN + n0 + it) * (3 * CC) + s * CC + h * DD + d];
      if (s == 0) val *= 0.125f;           // q * D^-0.5
      kqv[it][s][d] = val;
    }
    __syncthreads();

    #pragma unroll
    for (int it = 0; it < TOK_BLK; ++it) {
      const float* qt = kqv[it][0];
      const float* kt = kqv[it][1];
      const float* vt = kqv[it][2];
      float a0 = 0.f, a1 = 0.f;            // dual accumulators: halve chain depth
      #pragma unroll
      for (int i = 0; i < 8; i += 2) {
        a0 = fmaf(kt[kk0 + i],     S[i],     a0);
        a1 = fmaf(kt[kk0 + i + 1], S[i + 1], a1);
      }
      float ks = a0 + a1;
      ks += __shfl_xor(ks, 1, 32); ks += __shfl_xor(ks, 2, 32); ks += __shfl_xor(ks, 4, 32);
      float u = vt[v] - ks;                // beta = 1 exactly
      float o0 = 0.f, o1 = 0.f;
      #pragma unroll
      for (int i = 0; i < 8; i += 2) {
        S[i]     = fmaf(kt[kk0 + i],     u, S[i]);     o0 = fmaf(qt[kk0 + i],     S[i],     o0);
        S[i + 1] = fmaf(kt[kk0 + i + 1], u, S[i + 1]); o1 = fmaf(qt[kk0 + i + 1], S[i + 1], o1);
      }
      float o = o0 + o1;
      o += __shfl_xor(o, 1, 32); o += __shfl_xor(o, 2, 32); o += __shfl_xor(o, 4, 32);
      if (sub == 0)
        dout[(size_t)(b * NN + n0 + it) * CC + h * DD + v] = o;   // [B,N,H,D]
    }
    __syncthreads();
  }
}

// ---------------- fused FIR + mix + gates + prune + entropies ----------------
__global__ __launch_bounds__(256) void fuse_kernel(
    const float* __restrict__ qkv, const float* __restrict__ id_raw,
    const float* __restrict__ dout, const float* __restrict__ idg_log,
    const float* __restrict__ prune_log, const float* __restrict__ g_log,
    const float* __restrict__ fir_s, const float* __restrict__ fir_l,
    const float* __restrict__ mix_w, const float* __restrict__ id_static,
    float* __restrict__ gated, float* __restrict__ ent_out) {
  const int tok = blockIdx.x;
  const int b = tok >> 10, n = tok & 1023;
  const int t = threadIdx.x;
  __shared__ float fs[HH][DD], fl[HH][DD];
  __shared__ float red[256];

  // causal depthwise FIR taps
  #pragma unroll
  for (int i = 0; i < 4; ++i) {
    int e = t + 256 * i, h = e >> 6, d = e & 63;
    float s = 0.f, l = 0.f;
    #pragma unroll
    for (int kk = 0; kk < 3; ++kk) {
      int m = n - 2 + kk;
      if (m >= 0) s = fmaf(fir_s[(size_t)e * 3 + kk],
                           qkv[(size_t)(b * NN + m) * (3 * CC) + 2 * CC + e], s);
    }
    #pragma unroll
    for (int kk = 0; kk < 7; ++kk) {
      int m = n - 6 + kk;
      if (m >= 0) l = fmaf(fir_l[(size_t)e * 7 + kk],
                           qkv[(size_t)(b * NN + m) * (3 * CC) + 2 * CC + e], l);
    }
    fs[h][d] = s; fl[h][d] = l;
  }
  __syncthreads();

  float ent_local = 0.f;
  #pragma unroll
  for (int i = 0; i < 4; ++i) {
    int e = t + 256 * i, h = e >> 6, d = e & 63;
    // residual cross-head mix
    float ms = fs[h][d], ml = fl[h][d];
    #pragma unroll
    for (int hh = 0; hh < HH; ++hh) {
      float mw = mix_w[hh * HH + h];
      ms = fmaf(fs[hh][d], mw, ms);
      ml = fmaf(fl[hh][d], mw, ml);
    }
    float dl  = dout[(size_t)tok * CC + e];
    float vr  = qkv[(size_t)tok * (3 * CC) + 2 * CC + e];
    float idv = id_raw[(size_t)tok * CC + e];
    float gate_id = sigmoidf_(idg_log[tok * HH + h]) * sigmoidf_(id_static[h]);
    idv *= gate_id;

    // prune softmax over 5 paths
    const float* pl = prune_log + (size_t)tok * (HH * PP) + h * PP;
    float mx = pl[0];
    #pragma unroll
    for (int p = 1; p < PP; ++p) mx = fmaxf(mx, pl[p]);
    float w[PP], sum = 0.f;
    #pragma unroll
    for (int p = 0; p < PP; ++p) { w[p] = __expf(pl[p] - mx); sum += w[p]; }
    float inv = 1.0f / sum;
    #pragma unroll
    for (int p = 0; p < PP; ++p) w[p] *= inv;

    float pruned = w[0]*ms + w[1]*ml + w[2]*dl + w[3]*vr + w[4]*idv;
    float g = sigmoidf_(g_log[tok * HH + h]);
    gated[(size_t)tok * CC + e] = pruned * g;

    if (d == 0) {  // one entropy contribution per (token, head)
      float e1 = 0.f;
      #pragma unroll
      for (int p = 0; p < PP; ++p) e1 -= w[p] * __logf(w[p] + 1e-9f);
      float e2 = -(g * __logf(g + 1e-9f) + (1.f - g) * __logf(1.f - g + 1e-9f));
      ent_local += e1 + e2;
    }
  }
  red[t] = ent_local;
  __syncthreads();
  for (int s = 128; s > 0; s >>= 1) { if (t < s) red[t] += red[t + s]; __syncthreads(); }
  if (t == 0) atomicAdd(ent_out, red[0] * (1.0f / (float)(TOKS * HH)));
}

__global__ void zero_scalar(float* p) { if (threadIdx.x == 0) p[0] = 0.f; }

// ---------------- launcher ----------------
extern "C" void kernel_launch(void* const* d_in, const int* in_sizes, int n_in,
                              void* d_out, int out_size, void* d_ws, size_t ws_size,
                              hipStream_t stream) {
  (void)in_sizes; (void)n_in; (void)out_size; (void)ws_size;
  const float* x          = (const float*)d_in[0];
  const float* w_qkv      = (const float*)d_in[1];
  const float* w_proj     = (const float*)d_in[2];
  const float* b_proj     = (const float*)d_in[3];
  const float* w_id       = (const float*)d_in[4];
  const float* id_static  = (const float*)d_in[5];
  const float* w_id_gate  = (const float*)d_in[6];
  const float* b_id_gate  = (const float*)d_in[7];
  const float* fir_s      = (const float*)d_in[8];
  const float* fir_l      = (const float*)d_in[9];
  const float* mix_w      = (const float*)d_in[10];
  const float* w_prune    = (const float*)d_in[11];
  const float* w_g1       = (const float*)d_in[12];
  const float* b_g1       = (const float*)d_in[13];
  const float* w_g2       = (const float*)d_in[14];
  float* out = (float*)d_out;
  float* ws  = (float*)d_ws;

  // workspace layout (floats)
  float* qkvb   = ws;                            // 2048*3072
  float* idb    = ws + (size_t)6291456;          // 2048*1024
  float* h1     = ws + (size_t)8388608;          // 2048*1024 (reused as `gated`)
  float* deltab = ws + (size_t)10485760;         // 2048*1024
  float* idg    = ws + (size_t)12582912;         // 2048*16
  float* prune  = ws + (size_t)12615680;         // 2048*80
  float* glog   = ws + (size_t)12779520;         // 2048*16

  zero_scalar<<<1, 1, 0, stream>>>(out + OUT_ELEMS);

  // big WMMA GEMMs from x
  wmma_gemm<<<dim3(3 * CC / 64, TOKS / 64), 128, 0, stream>>>(x, w_qkv, nullptr, qkvb, TOKS, 3 * CC, CC, 0);
  wmma_gemm<<<dim3(CC / 64, TOKS / 64), 128, 0, stream>>>(x, w_id,  nullptr, idb, TOKS, CC, CC, 0);
  wmma_gemm<<<dim3(CC / 64, TOKS / 64), 128, 0, stream>>>(x, w_g1,  b_g1,    h1,  TOKS, CC, CC, 1);  // fused gelu

  // small projections
  small_gemm<<<(TOKS * HH) / 8, 256, 0, stream>>>(x,  w_id_gate, b_id_gate, idg,   TOKS, CC, HH);
  small_gemm<<<(TOKS * HH * PP) / 8, 256, 0, stream>>>(x, w_prune, nullptr, prune, TOKS, CC, HH * PP);
  small_gemm<<<(TOKS * HH) / 8, 256, 0, stream>>>(h1, w_g2, nullptr, glog, TOKS, CC, HH);

  // sequential delta recurrence (beta == 1), K-split across lanes
  delta_kernel<<<BB * HH, 512, 0, stream>>>(qkvb, deltab);

  // fused FIR/mix/gates/prune + entropies (writes `gated` into h1 buffer)
  fuse_kernel<<<TOKS, 256, 0, stream>>>(qkvb, idb, deltab, idg, prune, glog,
                                        fir_s, fir_l, mix_w, id_static,
                                        h1, out + OUT_ELEMS);

  // final projection -> out
  wmma_gemm<<<dim3(CC / 64, TOKS / 64), 128, 0, stream>>>(h1, w_proj, b_proj, out, TOKS, CC, CC, 0);
}
